// MoELayer_84413287236139
// MI455X (gfx1250) — matrix operand order repaired and checked by reading, not compiled, not run on hardware
//
#include <hip/hip_runtime.h>
#include <hip/hip_bf16.h>
#include <stdint.h>

#define DEV_INLINE __device__ __forceinline__

// Fixed problem sizes from the reference
constexpr int kD    = 1024;
constexpr int kH    = 4096;
constexpr int kE    = 8;
constexpr int kNT   = 4096;         // B*S tokens
constexpr int kTOPK = 2;
constexpr int kNA   = kNT * kTOPK;  // 8192 (token, expert) assignments

typedef __attribute__((ext_vector_type(16))) __bf16 v16bf;
typedef __attribute__((ext_vector_type(8)))  __bf16 v8bf;
typedef __attribute__((ext_vector_type(8)))  float  v8f;

constexpr int TM = 128, TN = 256, TK = 32;
constexpr int LDSK = TK + 8;  // 40 halves (80 B) per LDS row -> frag reads 16B aligned

DEV_INLINE v16bf cat16(v8bf lo, v8bf hi) {
  return __builtin_shufflevector(lo, hi, 0,1,2,3,4,5,6,7,8,9,10,11,12,13,14,15);
}

// A (16x32 bf16) fragment per ISA: lane row = lane&15;
// halves 0..7 -> K = kb..kb+7, halves 8..15 -> K = kb+16..kb+23, kb = 8*(lane>=16)
DEV_INLINE v16bf load_a_frag(const __bf16* As, int row0, int lane) {
  const int r  = row0 + (lane & 15);
  const int kb = (lane & 16) ? 8 : 0;
  v8bf lo = *(const v8bf*)(As + r * LDSK + kb);
  v8bf hi = *(const v8bf*)(As + r * LDSK + kb + 16);
  return cat16(lo, hi);
}

// B (32x16 bf16) fragment: lane col = lane&15; halves 0..7 -> K = kb..kb+7,
// halves 8..15 -> K = kb+16..kb+23, kb = 8*(lane>=16). Bs is stored transposed: Bs[n][k].
DEV_INLINE v16bf load_b_frag(const __bf16* Bs, int col0, int lane) {
  const int c  = col0 + (lane & 15);
  const int kb = (lane & 16) ? 8 : 0;
  v8bf lo = *(const v8bf*)(Bs + c * LDSK + kb);
  v8bf hi = *(const v8bf*)(Bs + c * LDSK + kb + 16);
  return cat16(lo, hi);
}

DEV_INLINE v8f wmma_bf16(v16bf a, v16bf b, v8f c) {
  return __builtin_amdgcn_wmma_f32_16x16x32_bf16(false, a, false, b, (short)0, c,
                                                 false, false);
}

// B tile staging: thread owns rows (bk2, bk2+1) x 16 cols; packs two K-adjacent
// bf16 into one b32 LDS store into the transposed Bs[n][k] layout.
DEV_INLINE void store_b_lds(__bf16* Bs, int bn, int bk2, const float4 (&vb)[8]) {
  float r0[16] = {vb[0].x, vb[0].y, vb[0].z, vb[0].w, vb[1].x, vb[1].y, vb[1].z, vb[1].w,
                  vb[2].x, vb[2].y, vb[2].z, vb[2].w, vb[3].x, vb[3].y, vb[3].z, vb[3].w};
  float r1[16] = {vb[4].x, vb[4].y, vb[4].z, vb[4].w, vb[5].x, vb[5].y, vb[5].z, vb[5].w,
                  vb[6].x, vb[6].y, vb[6].z, vb[6].w, vb[7].x, vb[7].y, vb[7].z, vb[7].w};
#pragma unroll
  for (int j = 0; j < 16; ++j) {
    union { __bf16 h[2]; uint32_t u; } pk;
    pk.h[0] = (__bf16)r0[j];
    pk.h[1] = (__bf16)r1[j];
    *(uint32_t*)(Bs + (bn + j) * LDSK + bk2) = pk.u;  // bk2 even -> 4B aligned
  }
}

DEV_INLINE void load_b_global(const float* brow0, const float* brow1, float4 (&vb)[8]) {
  const float4* p0 = (const float4*)brow0;
  const float4* p1 = (const float4*)brow1;
#pragma unroll
  for (int j = 0; j < 4; ++j) { vb[j] = p0[j]; vb[j + 4] = p1[j]; }
}

// ---------------------------------------------------------------- routing ---

__global__ void moe_init_kernel(int* counts, int* cursor) {
  const int t = threadIdx.x;
  if (t < kE) { counts[t] = 0; cursor[t] = 0; }
}

__global__ __launch_bounds__(256) void moe_gate_kernel(
    const float* __restrict__ x, const float* __restrict__ gw,
    const float* __restrict__ gb, int* __restrict__ topk_idx,
    float* __restrict__ topk_val, int* __restrict__ counts) {
  const int lane = threadIdx.x & 31;
  const int t = blockIdx.x * 8 + (threadIdx.x >> 5);  // one wave32 per token
  if (t >= kNT) return;
  const float* xr = x + (size_t)t * kD;
  float acc[kE];
#pragma unroll
  for (int e = 0; e < kE; ++e) acc[e] = 0.f;
  for (int d = lane; d < kD; d += 32) {
    const float xv = xr[d];
    const float4* g = (const float4*)(gw + (size_t)d * kE);
    const float4 g0 = g[0], g1 = g[1];
    acc[0] += xv * g0.x; acc[1] += xv * g0.y; acc[2] += xv * g0.z; acc[3] += xv * g0.w;
    acc[4] += xv * g1.x; acc[5] += xv * g1.y; acc[6] += xv * g1.z; acc[7] += xv * g1.w;
  }
#pragma unroll
  for (int e = 0; e < kE; ++e)
#pragma unroll
    for (int off = 16; off > 0; off >>= 1)
      acc[e] += __shfl_xor(acc[e], off, 32);
  if (lane == 0) {
    float lg[kE], m = -3.4e38f;
#pragma unroll
    for (int e = 0; e < kE; ++e) { lg[e] = acc[e] + gb[e]; m = fmaxf(m, lg[e]); }
    float p[kE], s = 0.f;
#pragma unroll
    for (int e = 0; e < kE; ++e) { p[e] = __expf(lg[e] - m); s += p[e]; }
    const float inv = 1.f / s;
#pragma unroll
    for (int e = 0; e < kE; ++e) p[e] *= inv;
    int i0 = 0;
#pragma unroll
    for (int e = 1; e < kE; ++e) if (p[e] > p[i0]) i0 = e;  // ties -> lowest idx
    int i1 = (i0 == 0) ? 1 : 0;
#pragma unroll
    for (int e = 0; e < kE; ++e) if (e != i0 && p[e] > p[i1]) i1 = e;
    topk_idx[t * 2 + 0] = i0; topk_val[t * 2 + 0] = p[i0];
    topk_idx[t * 2 + 1] = i1; topk_val[t * 2 + 1] = p[i1];
    atomicAdd(&counts[i0], 1);
    atomicAdd(&counts[i1], 1);
  }
}

__global__ void moe_offsets_kernel(const int* counts, int* offsets) {
  if (threadIdx.x == 0 && blockIdx.x == 0) {
    int a = 0;
    for (int e = 0; e < kE; ++e) { offsets[e] = a; a += counts[e]; }
    offsets[kE] = a;
  }
}

__global__ __launch_bounds__(256) void moe_scatter_kernel(
    const int* __restrict__ topk_idx, const int* __restrict__ offsets,
    int* __restrict__ cursor, int* __restrict__ assign_token,
    int* __restrict__ tok_slot) {
  const int t = blockIdx.x * blockDim.x + threadIdx.x;
  if (t >= kNT) return;
#pragma unroll
  for (int k = 0; k < kTOPK; ++k) {
    const int e = topk_idx[t * 2 + k];
    const int pos = atomicAdd(&cursor[e], 1);
    const int slot = offsets[e] + pos;
    assign_token[slot] = t;
    tok_slot[t * 2 + k] = slot;
  }
}

// -------------------------------------------------------------- FFN GEMM 1 --
// hidden[slot, H] = relu(x[token(slot), :] @ w1[e] + b1[e])   (bf16 out)
// 128x256 block tile, 8 waves as 2x4, 64x64 wave tile (16 WMMA / chunk),
// double-buffered LDS: one barrier per K-chunk, HBM latency behind WMMA.

__global__ __launch_bounds__(256) void moe_ffn1_kernel(
    const float* __restrict__ x, const float* __restrict__ w1,
    const float* __restrict__ b1, const int* __restrict__ assign_token,
    const int* __restrict__ offsets, __bf16* __restrict__ hidden) {
  __shared__ __align__(16) __bf16 As[2][TM * LDSK];
  __shared__ __align__(16) __bf16 Bs[2][TN * LDSK];

  const int e = blockIdx.z;
  const int base = offsets[e];
  const int rem = (offsets[e + 1] - base) - (int)blockIdx.y * TM;
  if (rem <= 0) return;
  const int n0 = blockIdx.x * TN;

  const int tid = threadIdx.x, lane = tid & 31, wave = tid >> 5;
  const int wm = (wave >> 2) * 64;  // 2 waves along M
  const int wn = (wave & 3) * 64;   // 4 waves along N, 64 cols each

  // A loader: 2 threads/row, 16 fp32 each
  const int ar = tid >> 1;
  const int ak = (tid & 1) * 16;
  const bool avalid = (ar < rem);
  const float* aptr = nullptr;
  if (avalid) {
    const int tok = assign_token[base + blockIdx.y * TM + ar];
    aptr = x + (size_t)tok * kD + ak;
  }
  // B loader: thread owns k-rows (bk2, bk2+1), 16 cols
  const int bk2 = (tid >> 4) << 1;
  const int bn  = (tid & 15) * 16;
  const float* bptr = w1 + (size_t)e * kD * kH + (size_t)bk2 * kH + n0 + bn;

  v8f acc[4][4];
#pragma unroll
  for (int mi = 0; mi < 4; ++mi)
#pragma unroll
    for (int ni = 0; ni < 4; ++ni)
#pragma unroll
      for (int r = 0; r < 8; ++r) acc[mi][ni][r] = 0.f;

  // ---- prologue: stage chunk 0 into buffer 0
  {
    float4 va[4];
    if (avalid) {
      const float4* p = (const float4*)aptr;
#pragma unroll
      for (int j = 0; j < 4; ++j) va[j] = p[j];
    } else {
#pragma unroll
      for (int j = 0; j < 4; ++j) va[j] = float4{0.f, 0.f, 0.f, 0.f};
    }
    float fa[16] = {va[0].x, va[0].y, va[0].z, va[0].w, va[1].x, va[1].y, va[1].z, va[1].w,
                    va[2].x, va[2].y, va[2].z, va[2].w, va[3].x, va[3].y, va[3].z, va[3].w};
    v8bf lo, hi;
#pragma unroll
    for (int j = 0; j < 8; ++j) { lo[j] = (__bf16)fa[j]; hi[j] = (__bf16)fa[j + 8]; }
    *(v8bf*)(As[0] + ar * LDSK + ak) = lo;
    *(v8bf*)(As[0] + ar * LDSK + ak + 8) = hi;

    float4 vb[8];
    load_b_global(bptr, bptr + kH, vb);
    store_b_lds(Bs[0], bn, bk2, vb);
  }
  __syncthreads();

  int cur = 0;
  for (int k0 = 0; k0 < kD; k0 += TK) {
    const bool has_next = (k0 + TK) < kD;

    // ---- issue next chunk's global loads (latency hidden behind WMMAs)
    float4 va[4], vb[8];
    if (has_next) {
      if (avalid) {
        const float4* p = (const float4*)(aptr + k0 + TK);
#pragma unroll
        for (int j = 0; j < 4; ++j) va[j] = p[j];
      } else {
#pragma unroll
        for (int j = 0; j < 4; ++j) va[j] = float4{0.f, 0.f, 0.f, 0.f};
      }
      load_b_global(bptr + (size_t)(k0 + TK) * kH,
                    bptr + (size_t)(k0 + TK + 1) * kH, vb);
      if (k0 + 2 * TK < kD)
        __builtin_prefetch(bptr + (size_t)(k0 + 2 * TK) * kH, 0, 1);
    }

    // ---- compute current chunk
    {
      const __bf16* Ac = As[cur];
      const __bf16* Bc = Bs[cur];
      v16bf af[4];
#pragma unroll
      for (int mi = 0; mi < 4; ++mi) af[mi] = load_a_frag(Ac, wm + mi * 16, lane);
#pragma unroll
      for (int ni = 0; ni < 4; ++ni) {
        const v16bf bfg = load_b_frag(Bc, wn + ni * 16, lane);
#pragma unroll
        for (int mi = 0; mi < 4; ++mi)
          acc[mi][ni] = wmma_bf16(af[mi], bfg, acc[mi][ni]);
      }
    }

    // ---- stage next chunk into the other buffer (no conflict with reads)
    if (has_next) {
      float fa[16] = {va[0].x, va[0].y, va[0].z, va[0].w, va[1].x, va[1].y, va[1].z, va[1].w,
                      va[2].x, va[2].y, va[2].z, va[2].w, va[3].x, va[3].y, va[3].z, va[3].w};
      v8bf lo, hi;
#pragma unroll
      for (int j = 0; j < 8; ++j) { lo[j] = (__bf16)fa[j]; hi[j] = (__bf16)fa[j + 8]; }
      *(v8bf*)(As[cur ^ 1] + ar * LDSK + ak) = lo;
      *(v8bf*)(As[cur ^ 1] + ar * LDSK + ak + 8) = hi;
      store_b_lds(Bs[cur ^ 1], bn, bk2, vb);
    }
    __syncthreads();
    cur ^= 1;
  }

  // ---- epilogue: bias + ReLU, bf16 rows of hidden
  const int mrow0 = base + blockIdx.y * TM;
#pragma unroll
  for (int ni = 0; ni < 4; ++ni) {
    const int col = n0 + wn + ni * 16 + (lane & 15);
    const float bias = b1[e * kH + col];
#pragma unroll
    for (int mi = 0; mi < 4; ++mi) {
      const int lr0 = wm + mi * 16 + ((lane & 16) ? 8 : 0);
#pragma unroll
      for (int r = 0; r < 8; ++r) {
        const int lr = lr0 + r;
        if (lr < rem) {
          const float h = acc[mi][ni][r] + bias;
          hidden[(size_t)(mrow0 + lr) * kH + col] = (__bf16)fmaxf(h, 0.f);
        }
      }
    }
  }
}

// -------------------------------------------------------------- FFN GEMM 2 --
// buf2[slot, D] = hidden[slot, :] @ w2[e] + b2[e]   (fp32 out)

__global__ __launch_bounds__(256) void moe_ffn2_kernel(
    const __bf16* __restrict__ hidden, const float* __restrict__ w2,
    const float* __restrict__ b2, const int* __restrict__ offsets,
    float* __restrict__ buf2) {
  __shared__ __align__(16) __bf16 As[2][TM * LDSK];
  __shared__ __align__(16) __bf16 Bs[2][TN * LDSK];

  const int e = blockIdx.z;
  const int base = offsets[e];
  const int rem = (offsets[e + 1] - base) - (int)blockIdx.y * TM;
  if (rem <= 0) return;
  const int n0 = blockIdx.x * TN;

  const int tid = threadIdx.x, lane = tid & 31, wave = tid >> 5;
  const int wm = (wave >> 2) * 64, wn = (wave & 3) * 64;

  const int ar = tid >> 1;
  const int ak = (tid & 1) * 16;
  const bool avalid = (ar < rem);
  const __bf16* aptr = nullptr;
  if (avalid) aptr = hidden + (size_t)(base + blockIdx.y * TM + ar) * kH + ak;

  const int bk2 = (tid >> 4) << 1;
  const int bn  = (tid & 15) * 16;
  const float* bptr = w2 + (size_t)e * kH * kD + (size_t)bk2 * kD + n0 + bn;

  v8f acc[4][4];
#pragma unroll
  for (int mi = 0; mi < 4; ++mi)
#pragma unroll
    for (int ni = 0; ni < 4; ++ni)
#pragma unroll
      for (int r = 0; r < 8; ++r) acc[mi][ni][r] = 0.f;

  // ---- prologue: chunk 0 -> buffer 0
  {
    v8bf lo, hi;
    if (avalid) {
      const v8bf* p = (const v8bf*)aptr;
      lo = p[0]; hi = p[1];
    } else {
#pragma unroll
      for (int j = 0; j < 8; ++j) { lo[j] = (__bf16)0.f; hi[j] = (__bf16)0.f; }
    }
    *(v8bf*)(As[0] + ar * LDSK + ak) = lo;
    *(v8bf*)(As[0] + ar * LDSK + ak + 8) = hi;

    float4 vb[8];
    load_b_global(bptr, bptr + kD, vb);
    store_b_lds(Bs[0], bn, bk2, vb);
  }
  __syncthreads();

  int cur = 0;
  for (int k0 = 0; k0 < kH; k0 += TK) {
    const bool has_next = (k0 + TK) < kH;

    v8bf lo, hi;
    float4 vb[8];
    if (has_next) {
      if (avalid) {
        const v8bf* p = (const v8bf*)(aptr + k0 + TK);
        lo = p[0]; hi = p[1];
      } else {
#pragma unroll
        for (int j = 0; j < 8; ++j) { lo[j] = (__bf16)0.f; hi[j] = (__bf16)0.f; }
      }
      load_b_global(bptr + (size_t)(k0 + TK) * kD,
                    bptr + (size_t)(k0 + TK + 1) * kD, vb);
      if (k0 + 2 * TK < kH)
        __builtin_prefetch(bptr + (size_t)(k0 + 2 * TK) * kD, 0, 1);
    }

    {
      const __bf16* Ac = As[cur];
      const __bf16* Bc = Bs[cur];
      v16bf af[4];
#pragma unroll
      for (int mi = 0; mi < 4; ++mi) af[mi] = load_a_frag(Ac, wm + mi * 16, lane);
#pragma unroll
      for (int ni = 0; ni < 4; ++ni) {
        const v16bf bfg = load_b_frag(Bc, wn + ni * 16, lane);
#pragma unroll
        for (int mi = 0; mi < 4; ++mi)
          acc[mi][ni] = wmma_bf16(af[mi], bfg, acc[mi][ni]);
      }
    }

    if (has_next) {
      *(v8bf*)(As[cur ^ 1] + ar * LDSK + ak) = lo;
      *(v8bf*)(As[cur ^ 1] + ar * LDSK + ak + 8) = hi;
      store_b_lds(Bs[cur ^ 1], bn, bk2, vb);
    }
    __syncthreads();
    cur ^= 1;
  }

  const int mrow0 = base + blockIdx.y * TM;
#pragma unroll
  for (int ni = 0; ni < 4; ++ni) {
    const int col = n0 + wn + ni * 16 + (lane & 15);
    const float bias = b2[e * kD + col];
#pragma unroll
    for (int mi = 0; mi < 4; ++mi) {
      const int lr0 = wm + mi * 16 + ((lane & 16) ? 8 : 0);
#pragma unroll
      for (int r = 0; r < 8; ++r) {
        const int lr = lr0 + r;
        if (lr < rem)
          buf2[(size_t)(mrow0 + lr) * kD + col] = acc[mi][ni][r] + bias;
      }
    }
  }
}

// ---------------------------------------------------------------- combine ---

__global__ __launch_bounds__(256) void moe_combine_kernel(
    const float* __restrict__ buf2, const int* __restrict__ tok_slot,
    const float* __restrict__ topk_val, float* __restrict__ out) {
  const int idx = blockIdx.x * blockDim.x + threadIdx.x;
  constexpr int DP = kD / 4;
  if (idx >= kNT * DP) return;
  const int t = idx / DP, d4 = idx - t * DP;
  const int s0 = tok_slot[t * 2 + 0], s1 = tok_slot[t * 2 + 1];
  const float v0 = topk_val[t * 2 + 0], v1 = topk_val[t * 2 + 1];
  const float4 a = ((const float4*)(buf2 + (size_t)s0 * kD))[d4];
  const float4 b = ((const float4*)(buf2 + (size_t)s1 * kD))[d4];
  float4 o;
  o.x = v0 * a.x + v1 * b.x;
  o.y = v0 * a.y + v1 * b.y;
  o.z = v0 * a.z + v1 * b.z;
  o.w = v0 * a.w + v1 * b.w;
  ((float4*)(out + (size_t)t * kD))[d4] = o;
}

// ----------------------------------------------------------------- launch ---

extern "C" void kernel_launch(void* const* d_in, const int* in_sizes, int n_in,
                              void* d_out, int out_size, void* d_ws, size_t ws_size,
                              hipStream_t stream) {
  (void)in_sizes; (void)n_in; (void)out_size; (void)ws_size;
  const float* x  = (const float*)d_in[0];
  const float* gw = (const float*)d_in[1];
  const float* gb = (const float*)d_in[2];
  const float* w1 = (const float*)d_in[3];
  const float* b1 = (const float*)d_in[4];
  const float* w2 = (const float*)d_in[5];
  const float* b2 = (const float*)d_in[6];
  float* out = (float*)d_out;

  char* ws = (char*)d_ws;
  constexpr size_t kHiddenBytes = (size_t)kNA * kH * 2;  // 64 MiB bf16
  constexpr size_t kBuf2Bytes   = (size_t)kNA * kD * 4;  // 32 MiB fp32
  __bf16* hidden = (__bf16*)ws;
  float*  buf2   = (float*)(ws + kHiddenBytes);
  char* meta = ws + kHiddenBytes + kBuf2Bytes;
  int*   topk_idx     = (int*)(meta + 0 * 32768);
  float* topk_val     = (float*)(meta + 1 * 32768);
  int*   assign_token = (int*)(meta + 2 * 32768);
  int*   tok_slot     = (int*)(meta + 3 * 32768);
  int*   counts       = (int*)(meta + 4 * 32768);
  int*   cursor       = counts + kE;
  int*   offsets      = cursor + kE;

  moe_init_kernel<<<1, 32, 0, stream>>>(counts, cursor);
  moe_gate_kernel<<<kNT / 8, 256, 0, stream>>>(x, gw, gb, topk_idx, topk_val, counts);
  moe_offsets_kernel<<<1, 1, 0, stream>>>(counts, offsets);
  moe_scatter_kernel<<<kNT / 256, 256, 0, stream>>>(topk_idx, offsets, cursor,
                                                    assign_token, tok_slot);

  dim3 g1(kH / TN, kNA / TM, kE);  // 16 x 64 x 8 (inactive M-tiles exit early)
  moe_ffn1_kernel<<<g1, 256, 0, stream>>>(x, w1, b1, assign_token, offsets, hidden);
  dim3 g2(kD / TN, kNA / TM, kE);  // 4 x 64 x 8
  moe_ffn2_kernel<<<g2, 256, 0, stream>>>(hidden, w2, b2, offsets, buf2);

  const int total4 = kNT * (kD / 4);
  moe_combine_kernel<<<(total4 + 255) / 256, 256, 0, stream>>>(buf2, tok_slot,
                                                               topk_val, out);
}
